// AutoEncoder_17214228922494
// MI455X (gfx1250) — matrix operand order repaired and checked by reading, not compiled, not run on hardware
//
#include <hip/hip_runtime.h>

// ---------------------------------------------------------------------------
// Problem dims (match reference)
// ---------------------------------------------------------------------------
#define TT   512   // seq len
#define BB   64    // batch
#define FF   256   // features / decoder hidden
#define EE   512   // encoder hidden

typedef __attribute__((ext_vector_type(16))) __bf16        bf16x16;
typedef __attribute__((ext_vector_type(8)))  float         f32x8;
typedef __attribute__((ext_vector_type(4)))  unsigned int  u32x4;
typedef __attribute__((ext_vector_type(4)))  float         f32x4;

// ---------------------------------------------------------------------------
// Small helpers
// ---------------------------------------------------------------------------
__device__ __forceinline__ unsigned short f32_to_bf16(float f) {
    unsigned int u = __float_as_uint(f);
    u += 0x7FFFu + ((u >> 16) & 1u);          // round-to-nearest-even
    return (unsigned short)(u >> 16);
}
__device__ __forceinline__ float bf16_to_f32(unsigned short h) {
    return __uint_as_float(((unsigned int)h) << 16);
}
__device__ __forceinline__ float dsigm(float x) { return 1.0f / (1.0f + __expf(-x)); }
__device__ __forceinline__ float dtanh(float x) { return 2.0f / (1.0f + __expf(-2.0f * x)) - 1.0f; }

union BFrag  { bf16x16 v; u32x4 q[2]; unsigned short s[16]; };
union CHalf  { u32x4 q; unsigned short s[8]; };
union CFull  { f32x4 q[2]; float s[8]; };

__device__ __forceinline__ f32x8 wmma_bf16(bf16x16 a, bf16x16 b, f32x8 c) {
    return __builtin_amdgcn_wmma_f32_16x16x32_bf16(false, a, false, b, (short)0, c,
                                                   false, false);
}

// B-fragment (32x16, bf16) prepacked contiguously: 32 bytes per lane.
__device__ __forceinline__ bf16x16 load_frag32(const unsigned short* __restrict__ p) {
    BFrag r;
    r.q[0] = *(const u32x4*)(p);
    r.q[1] = *(const u32x4*)(p + 8);
    return r.v;
}

// A-fragment (16x32, bf16) from a row-major bf16 matrix (global memory).
// ISA layout: lanes 0-15 -> K 0..7 & 16..23 ; lanes 16-31 -> K 8..15 & 24..31.
__device__ __forceinline__ bf16x16 load_a_frag_global(const unsigned short* __restrict__ tile,
                                                      int stride, int lane) {
    const int m  = lane & 15;
    const int hs = (lane >> 4) & 1;
    const unsigned short* p = tile + m * stride + hs * 8;
    BFrag r;
    r.q[0] = *(const u32x4*)(p);
    r.q[1] = *(const u32x4*)(p + 16);
    return r.v;
}

// A-fragment from LDS with XOR chunk swizzle (chunk = 8 halves = 16B) so the
// 16 row-parallel lanes hit all 64 banks instead of one.
template <int RS, int CM>
__device__ __forceinline__ bf16x16 load_a_frag_lds(const unsigned short* __restrict__ lds,
                                                   int row0, int kbase, int lane) {
    const int m  = row0 + (lane & 15);
    const int hs = (lane >> 4) & 1;
    const int k0 = kbase + hs * 8;
    const int k1 = k0 + 16;
    const int c0 = (((k0 >> 3) ^ (m & CM)) << 3);
    const int c1 = (((k1 >> 3) ^ (m & CM)) << 3);
    BFrag r;
    r.q[0] = *(const u32x4*)(lds + m * RS + c0);
    r.q[1] = *(const u32x4*)(lds + m * RS + c1);
    return r.v;
}

// ---------------------------------------------------------------------------
// Prep kernels
// ---------------------------------------------------------------------------
__global__ void k_pack_bf16(const float* __restrict__ src, unsigned short* __restrict__ dst, int n) {
    int i = blockIdx.x * 256 + threadIdx.x;
    if (i < n) dst[i] = f32_to_bf16(src[i]);
}

// W is row-major (N x K) f32; build B-fragments for gates = act @ W^T.
__global__ void k_pack_bfrags(const float* __restrict__ W, unsigned short* __restrict__ dst,
                              int K, int KT, int n) {
    int i = blockIdx.x * 256 + threadIdx.x;
    if (i >= n) return;
    const int e    = i & 15;
    const int lane = (i >> 4) & 31;
    const int fi   = i >> 9;
    const int kt   = fi % KT;
    const int nt   = fi / KT;
    const int k  = kt * 32 + ((lane >> 4) & 1) * 8 + (e & 7) + (e >> 3) * 16;
    const int nn = nt * 16 + (lane & 15);
    dst[i] = f32_to_bf16(W[(size_t)nn * K + k]);
}

// xproj = X(32768 x 256) @ Wih^T + b -> (32768 x 2048), bf16 C-fragments.
__global__ void k_xproj(const unsigned short* __restrict__ X,
                        const unsigned short* __restrict__ wfrags,
                        const float* __restrict__ bias,
                        unsigned short* __restrict__ dst) {
    const int lane = threadIdx.x & 31;
    const int wg   = blockIdx.x * (blockDim.x >> 5) + (threadIdx.x >> 5);
    const int NT = 128, KT = 8;
    const int mt = wg / NT, nt = wg % NT;
    f32x8 acc = {};
    const unsigned short* at = X + (size_t)(mt * 16) * FF;
    for (int kt = 0; kt < KT; ++kt) {
        bf16x16 a = load_a_frag_global(at + kt * 32, FF, lane);
        bf16x16 b = load_frag32(wfrags + ((size_t)(nt * KT + kt)) * 512 + lane * 16);
        acc = wmma_bf16(a, b, acc);
    }
    const float bv = bias[nt * 16 + (lane & 15)];
    CHalf o;
#pragma unroll
    for (int i = 0; i < 8; ++i) o.s[i] = f32_to_bf16(acc[i] + bv);
    *(u32x4*)(dst + (size_t)wg * 256 + lane * 8) = o.q;
}

// ---------------------------------------------------------------------------
// Encoder recurrent scan, split over 4 WGPs per direction.
// Grid: 8 blocks x 256 threads. blockIdx>>2 = direction, blockIdx&3 = slice.
// Each wave owns a 16-column slice of h; c stays in VGPRs; h lives in a
// double-buffered global buffer (L2-resident) shared by the 4 workgroups,
// synchronized once per step with a monotonic atomic barrier + agent fences
// (acquire fence lowers to global_inv -> flushes stale WGP$ lines).
// ---------------------------------------------------------------------------
__global__ __launch_bounds__(256, 1)
void k_enc_scan(const unsigned short* __restrict__ whhF,
                const unsigned short* __restrict__ whhB,
                const unsigned short* __restrict__ xpF,
                const unsigned short* __restrict__ xpB,
                unsigned short* __restrict__ hg,      // [dir][2][64*512] bf16
                unsigned int*   __restrict__ cnt,     // [dir] barrier counters
                unsigned short* __restrict__ x0) {
    const int dir = blockIdx.x >> 2;
    const int sub = blockIdx.x & 3;
    const unsigned short* whh = dir ? whhB : whhF;
    const unsigned short* xp  = dir ? xpB : xpF;
    unsigned short* hbase = hg + (size_t)dir * 2 * (BB * EE);
    unsigned int*   bar   = cnt + dir;

    const int tid  = threadIdx.x;
    const int wave = tid >> 5, lane = tid & 31;
    const int lm = lane & 15, hs = (lane >> 4) & 1;
    const int cb = sub * 8 + wave;                 // column block 0..31
    const int j  = cb * 16 + lm;                   // my output column

    float cst[4][8], hnew[4][8];
#pragma unroll
    for (int a = 0; a < 4; ++a)
#pragma unroll
        for (int i = 0; i < 8; ++i) cst[a][i] = 0.f;

    for (int t = 0; t < TT; ++t) {
        const int xt = dir ? (TT - 1 - t) : t;
        const unsigned short* hcur = hbase + (size_t)(t & 1) * (BB * EE);
        unsigned short*       hnxt = hbase + (size_t)((t + 1) & 1) * (BB * EE);

        // Prefetch next step's xproj tiles HBM -> L2 (survives global_inv).
        if (t + 1 < TT) {
            const int xn = dir ? (TT - 2 - t) : (t + 1);
#pragma unroll
            for (int mt = 0; mt < 4; ++mt)
#pragma unroll
                for (int g = 0; g < 4; ++g)
                    __builtin_prefetch(xp + (((size_t)(xn * 4 + mt)) * 128 +
                                             (g * 32 + cb)) * 256 + lane * 8, 0, 1);
        }

#pragma unroll
        for (int mt = 0; mt < 4; ++mt) {
            // Issue xproj loads early; consumed only after the K loop so the
            // HBM/L2 latency is hidden behind 64 WMMAs.
            CHalf xv[4];
#pragma unroll
            for (int g = 0; g < 4; ++g)
                xv[g].q = *(const u32x4*)(xp + (((size_t)(xt * 4 + mt)) * 128 +
                                                (g * 32 + cb)) * 256 + lane * 8);
            f32x8 acc[4];
#pragma unroll
            for (int g = 0; g < 4; ++g) acc[g] = (f32x8){};
            for (int kt = 0; kt < 16; ++kt) {      // K = 512
                bf16x16 a = load_a_frag_global(hcur + (size_t)(mt * 16) * EE + kt * 32,
                                               EE, lane);
#pragma unroll
                for (int g = 0; g < 4; ++g) {
                    bf16x16 b = load_frag32(whh +
                        ((size_t)((g * 32 + cb) * 16 + kt)) * 512 + lane * 16);
                    acc[g] = wmma_bf16(a, b, acc[g]);
                }
            }
#pragma unroll
            for (int i = 0; i < 8; ++i) {
                float iv = dsigm(acc[0][i] + bf16_to_f32(xv[0].s[i]));
                float fv = dsigm(acc[1][i] + bf16_to_f32(xv[1].s[i]));
                float gv = dtanh(acc[2][i] + bf16_to_f32(xv[2].s[i]));
                float ov = dsigm(acc[3][i] + bf16_to_f32(xv[3].s[i]));
                float c  = fv * cst[mt][i] + iv * gv;
                cst[mt][i]  = c;
                hnew[mt][i] = ov * dtanh(c);
            }
        }
        // Publish my h slice for the next step.
#pragma unroll
        for (int mt = 0; mt < 4; ++mt)
#pragma unroll
            for (int i = 0; i < 8; ++i) {
                const int r = mt * 16 + i + 8 * hs;
                hnxt[(size_t)r * EE + j] = f32_to_bf16(hnew[mt][i]);
            }
        // Cross-WGP step barrier (monotonic counter, no reset).
        __builtin_amdgcn_fence(__ATOMIC_RELEASE, "agent");   // stores -> L2
        __syncthreads();
        if (tid == 0) {
            __hip_atomic_fetch_add(bar, 1u, __ATOMIC_RELAXED, __HIP_MEMORY_SCOPE_AGENT);
            const unsigned int want = 4u * (unsigned)(t + 1);
            while (__hip_atomic_load(bar, __ATOMIC_RELAXED, __HIP_MEMORY_SCOPE_AGENT) < want)
                __builtin_amdgcn_s_sleep(2);
        }
        __syncthreads();
        __builtin_amdgcn_fence(__ATOMIC_ACQUIRE, "agent");   // global_inv WGP$
    }
    // Final hidden lives in hg buffer 0 (TT even). One WG per dir exports it
    // into x0 (row-major bf16, stride 2E; fwd at 0, bwd at +E).
    if (sub == 0) {
        for (int i = tid; i < BB * EE; i += 256) {
            const int r = i >> 9, jj = i & (EE - 1);
            x0[(size_t)r * (2 * EE) + dir * EE + jj] = hbase[(size_t)r * EE + jj];
        }
    }
}

// ---------------------------------------------------------------------------
// Decoder: 1 block, 512 threads = 16 waves (per-step GEMM is 4x smaller, so a
// cross-WGP barrier would cost more than it saves). Prologue builds
// xp0 = x0@Wih^T + b via WMMA; scan keeps h in swizzled LDS, c in VGPRs.
// ---------------------------------------------------------------------------
__global__ __launch_bounds__(512, 1)
void k_decoder(const unsigned short* __restrict__ x0,
               const unsigned short* __restrict__ dwih,
               const float* __restrict__ db,
               const unsigned short* __restrict__ dwhh,
               float* __restrict__ xp0,
               float* __restrict__ out) {
    __shared__ unsigned short hlds[BB * FF];      // 32 KB bf16
    const int tid = threadIdx.x;
    for (int i = tid; i < BB * FF; i += 512) hlds[i] = 0;
    const int wave = tid >> 5, lane = tid & 31;
    const int lm = lane & 15, hs = (lane >> 4) & 1;

    // ---- phase 1: xp0 (64 x 1024) = x0 (64 x 1024) @ dec_Wih^T + dec_b ----
    for (int it = 0; it < 16; ++it) {
        const int tile = it * 16 + wave;          // 256 tiles: 4 mt x 64 nt
        const int mt = tile >> 6, nt = tile & 63;
        f32x8 acc = {};
        for (int kt = 0; kt < 32; ++kt) {         // K = 1024
            bf16x16 a = load_a_frag_global(x0 + (size_t)(mt * 16) * 1024 + kt * 32,
                                           1024, lane);
            bf16x16 b = load_frag32(dwih + ((size_t)(nt * 32 + kt)) * 512 + lane * 16);
            acc = wmma_bf16(a, b, acc);
        }
        const float bv = db[nt * 16 + lm];
        CFull o;
#pragma unroll
        for (int i = 0; i < 8; ++i) o.s[i] = acc[i] + bv;
        f32x4* d = (f32x4*)(xp0 + (size_t)tile * 256 + lane * 8);
        d[0] = o.q[0];
        d[1] = o.q[1];
    }
    __threadfence();
    __syncthreads();

    // ---- phase 2: recurrence ----
    float cst[4][8], hnew[4][8];
#pragma unroll
    for (int a = 0; a < 4; ++a)
#pragma unroll
        for (int i = 0; i < 8; ++i) cst[a][i] = 0.f;
    const int j = wave * 16 + lm;

    for (int t = 0; t < TT; ++t) {
#pragma unroll
        for (int mt = 0; mt < 4; ++mt) {
            // Constant input projection: issue loads early, add after K loop.
            CFull xv[4];
#pragma unroll
            for (int g = 0; g < 4; ++g) {
                const f32x4* p = (const f32x4*)(xp0 +
                    ((size_t)(mt * 64 + g * 16 + wave)) * 256 + lane * 8);
                xv[g].q[0] = p[0];
                xv[g].q[1] = p[1];
            }
            f32x8 acc[4];
#pragma unroll
            for (int g = 0; g < 4; ++g) acc[g] = (f32x8){};
            for (int kt = 0; kt < 8; ++kt) {      // K = 256
                bf16x16 a = load_a_frag_lds<FF, 31>(hlds, mt * 16, kt * 32, lane);
#pragma unroll
                for (int g = 0; g < 4; ++g) {
                    bf16x16 b = load_frag32(dwhh +
                        ((size_t)((g * 16 + wave) * 8 + kt)) * 512 + lane * 16);
                    acc[g] = wmma_bf16(a, b, acc[g]);
                }
            }
#pragma unroll
            for (int i = 0; i < 8; ++i) {
                float iv = dsigm(acc[0][i] + xv[0].s[i]);
                float fv = dsigm(acc[1][i] + xv[1].s[i]);
                float gv = dtanh(acc[2][i] + xv[2].s[i]);
                float ov = dsigm(acc[3][i] + xv[3].s[i]);
                float c  = fv * cst[mt][i] + iv * gv;
                cst[mt][i]  = c;
                hnew[mt][i] = ov * dtanh(c);
            }
        }
        __syncthreads();
#pragma unroll
        for (int mt = 0; mt < 4; ++mt)
#pragma unroll
            for (int i = 0; i < 8; ++i) {
                const int r  = mt * 16 + i + 8 * hs;   // batch row
                const int ch = (((j >> 3) ^ (r & 31)) << 3);
                hlds[r * FF + ch + (j & 7)] = f32_to_bf16(hnew[mt][i]);
                out[((size_t)t * BB + r) * FF + j] = hnew[mt][i];
            }
        __syncthreads();
    }
}

// ---------------------------------------------------------------------------
// Host side
// ---------------------------------------------------------------------------
extern "C" void kernel_launch(void* const* d_in, const int* in_sizes, int n_in,
                              void* d_out, int out_size, void* d_ws, size_t ws_size,
                              hipStream_t stream) {
    (void)in_sizes; (void)n_in; (void)out_size; (void)ws_size;
    const float* seq   = (const float*)d_in[0];
    const float* wihf  = (const float*)d_in[1];
    const float* whhf  = (const float*)d_in[2];
    const float* bf    = (const float*)d_in[3];
    const float* wihb  = (const float*)d_in[4];
    const float* whhb  = (const float*)d_in[5];
    const float* bb    = (const float*)d_in[6];
    const float* dwih  = (const float*)d_in[7];
    const float* dwhh  = (const float*)d_in[8];
    const float* db    = (const float*)d_in[9];
    float* out = (float*)d_out;

    char* ws = (char*)d_ws;
    size_t off = 0;
    auto alloc = [&](size_t bytes) -> void* {
        off = (off + 255) & ~(size_t)255;
        void* p = ws + off;
        off += bytes;
        return p;
    };
    unsigned short* X    = (unsigned short*)alloc((size_t)TT * BB * FF * 2);   // 16 MB
    unsigned short* WIHF = (unsigned short*)alloc((size_t)2048 * 256 * 2);
    unsigned short* WIHB = (unsigned short*)alloc((size_t)2048 * 256 * 2);
    unsigned short* WHHF = (unsigned short*)alloc((size_t)2048 * 512 * 2);
    unsigned short* WHHB = (unsigned short*)alloc((size_t)2048 * 512 * 2);
    unsigned short* DWIH = (unsigned short*)alloc((size_t)1024 * 1024 * 2);
    unsigned short* DWHH = (unsigned short*)alloc((size_t)1024 * 256 * 2);
    unsigned short* XPF  = (unsigned short*)alloc((size_t)TT * BB * 2048 * 2); // 128 MB
    unsigned short* XPB  = (unsigned short*)alloc((size_t)TT * BB * 2048 * 2); // 128 MB
    unsigned short* X0   = (unsigned short*)alloc((size_t)BB * 1024 * 2);
    float*          XP0  = (float*)         alloc((size_t)BB * 1024 * 4);
    unsigned short* HG   = (unsigned short*)alloc((size_t)2 * 2 * BB * EE * 2); // 256 KB
    unsigned int*   CNT  = (unsigned int*)  alloc(256);

    // Zero recurrent-state buffers and barrier counters (replayed every call).
    hipMemsetAsync(HG, 0, (size_t)2 * 2 * BB * EE * 2, stream);
    hipMemsetAsync(CNT, 0, 256, stream);

    // 1) cast activations to bf16
    k_pack_bf16<<<(TT * BB * FF) / 256, 256, 0, stream>>>(seq, X, TT * BB * FF);
    // 2) prepack all weight matrices into WMMA B-fragments (bf16)
    k_pack_bfrags<<<(2048 * 256) / 256, 256, 0, stream>>>(wihf, WIHF, 256, 8,  2048 * 256);
    k_pack_bfrags<<<(2048 * 256) / 256, 256, 0, stream>>>(wihb, WIHB, 256, 8,  2048 * 256);
    k_pack_bfrags<<<(2048 * 512) / 256, 256, 0, stream>>>(whhf, WHHF, 512, 16, 2048 * 512);
    k_pack_bfrags<<<(2048 * 512) / 256, 256, 0, stream>>>(whhb, WHHB, 512, 16, 2048 * 512);
    k_pack_bfrags<<<(1024 * 1024) / 256, 256, 0, stream>>>(dwih, DWIH, 1024, 32, 1024 * 1024);
    k_pack_bfrags<<<(1024 * 256) / 256, 256, 0, stream>>>(dwhh, DWHH, 256, 8,  1024 * 256);
    // 3) whole-chip GEMMs: input projections for both directions
    k_xproj<<<32768, 256, 0, stream>>>(X, WIHF, bf, XPF);
    k_xproj<<<32768, 256, 0, stream>>>(X, WIHB, bb, XPB);
    // 4) encoder recurrent scans: 2 dirs x 4 WGPs, step-synced via atomics
    k_enc_scan<<<8, 256, 0, stream>>>(WHHF, WHHB, XPF, XPB, HG, CNT, X0);
    // 5) decoder (prologue GEMM + 512-step recurrence, writes preds)
    k_decoder<<<1, 512, 0, stream>>>(X0, DWIH, db, DWHH, XP0, out);
}
